// kNN_1717986918440
// MI455X (gfx1250) — compile-verified, matrix operand order checked by reading, not run on hardware
//
#include <hip/hip_runtime.h>
#include <hip/hip_bf16.h>
#include <float.h>

#define NUM_CLASSES 20
#define KNN 5
#define CFEAT 256
#define MMEM 10000
#define NPTS 32768      // 2*128*128
#define HW 16384        // 128*128
#define MCHUNKS 625     // 10000 / 16
#define CHUNK_BYTES (16 * CFEAT * 2)   // 8 KB of bf16

typedef __attribute__((ext_vector_type(16))) __bf16 v16bf;
typedef __attribute__((ext_vector_type(8)))  float  v8f;
typedef unsigned int v4u __attribute__((ext_vector_type(4)));
typedef int          v8i __attribute__((ext_vector_type(8)));
typedef int          v4i __attribute__((ext_vector_type(4)));

union Frag16 { v16bf v; uint4 u[2]; };

#if defined(__gfx1250__) && __has_builtin(__builtin_amdgcn_tensor_load_to_lds)
#define USE_TDM 1
#else
#define USE_TDM 0
#endif

// ---------------- phase 0a: x (b,c,h,w) f32 -> xb[n][c] bf16 (row major) ----
__global__ void convert_x_kernel(const float* __restrict__ x,
                                 __hip_bfloat16* __restrict__ xb) {
    unsigned t = blockIdx.x * blockDim.x + threadIdx.x;   // 8388608 total
    int b  = (int)(t >> 22);            // / (256*16384)
    int c  = (int)((t >> 14) & 255);
    int hw = (int)(t & 16383);
    int n  = b * HW + hw;
    xb[(size_t)n * CFEAT + c] = __float2bfloat16(x[t]);
}

// ---------------- phase 0b: memory f32 -> bf16(-2*m) -----------------------
__global__ void convert_mem_kernel(const float* __restrict__ mem,
                                   __hip_bfloat16* __restrict__ mb) {
    unsigned t = blockIdx.x * blockDim.x + threadIdx.x;   // 2,560,000 total
    mb[t] = __float2bfloat16(-2.0f * mem[t]);
}

// ---------------- phase 0c: ||m||^2 per row (one wave32 per row) -----------
__global__ void mem_norm_kernel(const float* __restrict__ mem,
                                float* __restrict__ mnorm) {
    int row  = blockIdx.x;
    int lane = threadIdx.x;
    const float* r = mem + (size_t)row * CFEAT;
    float s = 0.f;
    #pragma unroll
    for (int k = 0; k < CFEAT / 32; ++k) {
        float v = r[lane + 32 * k];
        s += v * v;
    }
    #pragma unroll
    for (int off = 16; off > 0; off >>= 1) s += __shfl_down(s, off, 32);
    if (lane == 0) mnorm[row] = s;
}

// ---- top-5 sorted insert (per lane) ----
__device__ __forceinline__ void insert5(float (&d5)[KNN], int (&i5)[KNN],
                                        float d, int j) {
    if (d < d5[KNN - 1]) {
        d5[KNN - 1] = d; i5[KNN - 1] = j;
        #pragma unroll
        for (int q = KNN - 1; q > 0; --q) {
            if (d5[q] < d5[q - 1]) {
                float td = d5[q]; d5[q] = d5[q-1]; d5[q-1] = td;
                int   ti = i5[q]; i5[q] = i5[q-1]; i5[q-1] = ti;
            }
        }
    }
}

// Stage one 16x256 bf16 memory chunk into LDS buffer `buf`.
__device__ __forceinline__ void stage_chunk(const __hip_bfloat16* mb, int chunk,
                                            unsigned char* smem_base, int buf) {
#if USE_TDM
    if ((threadIdx.x >> 5) == 0) {   // wave 0 drives the Tensor Data Mover
        unsigned long long ga =
            (unsigned long long)(uintptr_t)(mb + (size_t)chunk * 16 * CFEAT);
        // D# group 0: count=1 | lds_addr | global_addr[56:0] | type=2
        v4u g0 = { 0x1u,
                   (unsigned)(buf * CHUNK_BYTES),
                   (unsigned)ga,
                   ((unsigned)(ga >> 32) & 0x01FFFFFFu) | 0x80000000u };
        // D# group 1: data_size=2B; tensor 256x16; tile 256x16; stride0=256
        v8i g1 = { 0x00010000,          // data_size=1 (2 bytes)
                   0x01000000,          // tensor_dim0 = 256  (bits 79:48)
                   0x00100000,          // tensor_dim1 = 16   (bits 111:80)
                   0x01000000,          // tile_dim0  = 256   (bits 127:112)
                   16,                  // tile_dim1  = 16    (bits 143:128)
                   256,                 // tensor_dim0_stride (bits 207:160)
                   0x10000000,          // tensor_dim1_stride = 4096
                   0 };
        v4i gz4 = { 0, 0, 0, 0 };       // groups 2/3 unused (2D tensor)
        v8i gz8 = { 0, 0, 0, 0, 0, 0, 0, 0 };
        __builtin_amdgcn_tensor_load_to_lds(g0, g1, gz4, gz4, gz8, 0);
    }
#else
    // Fallback: cooperative copy, 256 threads x 2 x 16B = 8 KB
    const uint4* src = (const uint4*)(mb + (size_t)chunk * 16 * CFEAT);
    uint4* dst = (uint4*)(smem_base + buf * CHUNK_BYTES);
    dst[threadIdx.x]       = src[threadIdx.x];
    dst[threadIdx.x + 256] = src[threadIdx.x + 256];
#endif
}

// ---------------- fused GEMM (bf16 WMMA) + streaming top-5 + vote ----------
// Block = 256 threads = 8 waves; each wave owns TWO 16-point tiles (register
// blocking) -> block covers 256 points. Memory chunks staged once per block
// into double-buffered LDS by the TDM and consumed by all 8 waves.
__global__ __launch_bounds__(256) void knn_wmma_kernel(
        const __hip_bfloat16* __restrict__ xb,    // [NPTS][256] bf16
        const __hip_bfloat16* __restrict__ mb,    // [MMEM][256] bf16, scaled -2
        const float* __restrict__ mnorm,          // [MMEM] f32
        const int*   __restrict__ memy,           // [MMEM] labels
        float* __restrict__ out) {                // [2][20][128][128]
    __shared__ __align__(16) unsigned char smem[2 * CHUNK_BYTES];

    const int lane = threadIdx.x & 31;
    const int wv   = threadIdx.x >> 5;         // wave in block: 0..7
    const int col  = lane & 15;                // matrix column within tile
    const int half = lane >> 4;                // lane half selects K/M groups
    // two point tiles per wave
    const int tile0 = blockIdx.x * 16 + wv * 2;
    const int p0    = tile0 * 16 + col;
    const int p1    = p0 + 16;

    // Preload both B operands (32 points x K=256) into VGPRs.
    Frag16 B0[8], B1[8];
    {
        const __hip_bfloat16* r0 = xb + (size_t)p0 * CFEAT + half * 8;
        const __hip_bfloat16* r1 = xb + (size_t)p1 * CFEAT + half * 8;
        #pragma unroll
        for (int kc = 0; kc < 8; ++kc) {
            B0[kc].u[0] = *(const uint4*)(r0 + kc * 32);
            B0[kc].u[1] = *(const uint4*)(r0 + kc * 32 + 16);
            B1[kc].u[0] = *(const uint4*)(r1 + kc * 32);
            B1[kc].u[1] = *(const uint4*)(r1 + kc * 32 + 16);
        }
    }

    float d5a[KNN], d5b[KNN];
    int   i5a[KNN], i5b[KNN];
    #pragma unroll
    for (int i = 0; i < KNN; ++i) {
        d5a[i] = FLT_MAX; i5a[i] = 0;
        d5b[i] = FLT_MAX; i5b[i] = 0;
    }

    // Prologue: stage chunk 0 into buffer 0.
    stage_chunk(mb, 0, smem, 0);

    for (int c = 0; c < MCHUNKS; ++c) {
        const int cur = c & 1;
#if USE_TDM
        if (wv == 0) __builtin_amdgcn_s_wait_tensorcnt(0);
#endif
        __syncthreads();                       // buf[cur] ready for all waves
        if (c + 1 < MCHUNKS) stage_chunk(mb, c + 1, smem, cur ^ 1);

        // A fragments from LDS: row = col, K pattern as WMMA 16-bit layout.
        const unsigned char* abase =
            smem + cur * CHUNK_BYTES + (col * CFEAT + half * 8) * 2;
        v8f acc0 = {}, acc1 = {};
        #pragma unroll
        for (int kc = 0; kc < 8; ++kc) {
            Frag16 A;
            A.u[0] = *(const uint4*)(abase + kc * 64);
            A.u[1] = *(const uint4*)(abase + kc * 64 + 32);
            acc0 = __builtin_amdgcn_wmma_f32_16x16x32_bf16(
                       false, A.v, false, B0[kc].v, (short)0, acc0, false, false);
            acc1 = __builtin_amdgcn_wmma_f32_16x16x32_bf16(
                       false, A.v, false, B1[kc].v, (short)0, acc1, false, false);
        }

        // distances: s = ||m_j||^2 + (-2 <x,m_j>), j = c*16 + half*8 + v
        const float4* mn = (const float4*)(mnorm + c * 16 + half * 8);
        float4 mn0 = mn[0], mn1 = mn[1];
        float mnv[8] = { mn0.x, mn0.y, mn0.z, mn0.w, mn1.x, mn1.y, mn1.z, mn1.w };
        const int jbase = c * 16 + half * 8;
        #pragma unroll
        for (int v = 0; v < 8; ++v) {
            insert5(d5a, i5a, mnv[v] + acc0[v], jbase + v);
            insert5(d5b, i5b, mnv[v] + acc1[v], jbase + v);
        }
        __syncthreads();                       // all waves done with buf[cur]
    }

    // Merge half-wave candidate streams: lanes 0-15 pull from lane+16.
    #pragma unroll
    for (int i = 0; i < KNN; ++i) {
        float pa = __shfl(d5a[i], (lane & 15) + 16, 32);
        int   qa = __shfl(i5a[i], (lane & 15) + 16, 32);
        insert5(d5a, i5a, pa, qa);
        float pb = __shfl(d5b[i], (lane & 15) + 16, 32);
        int   qb = __shfl(i5b[i], (lane & 15) + 16, 32);
        insert5(d5b, i5b, pb, qb);
    }

    // Mode vote (ties -> smallest class) and one-hot write, per tile.
    #pragma unroll
    for (int t = 0; t < 2; ++t) {
        int   (&i5)[KNN] = t ? i5b : i5a;
        int lab[KNN];
        #pragma unroll
        for (int i = 0; i < KNN; ++i) lab[i] = memy[i5[i]];
        int best = NUM_CLASSES, bestc = 0;
        #pragma unroll
        for (int i = 0; i < KNN; ++i) {
            int cnt = 0;
            #pragma unroll
            for (int q = 0; q < KNN; ++q) cnt += (lab[q] == lab[i]) ? 1 : 0;
            if (cnt > bestc || (cnt == bestc && lab[i] < best)) { bestc = cnt; best = lab[i]; }
        }
        if (half == 0) {
            int p  = t ? p1 : p0;
            int b  = p >> 14;
            int hw = p & 16383;
            float* obase = out + (size_t)b * NUM_CLASSES * HW + hw;
            #pragma unroll
            for (int cls = 0; cls < NUM_CLASSES; ++cls)
                obase[(size_t)cls * HW] = (cls == best) ? 1.0f : 0.0f;
        }
    }
}

extern "C" void kernel_launch(void* const* d_in, const int* in_sizes, int n_in,
                              void* d_out, int out_size, void* d_ws, size_t ws_size,
                              hipStream_t stream) {
    const float* x    = (const float*)d_in[0];   // (2,256,128,128)
    // d_in[1] = y, unused by the reference eval path
    const float* memx = (const float*)d_in[2];   // (10000,256)
    const int*   memy = (const int*)d_in[3];     // (10000,)
    float* out = (float*)d_out;                  // (2,20,128,128)

    char* ws = (char*)d_ws;
    __hip_bfloat16* xb    = (__hip_bfloat16*)ws;                       // 16 MB
    __hip_bfloat16* mbuf  = (__hip_bfloat16*)(ws + 16777216);          // 5 MB
    float*          mnorm = (float*)(ws + 16777216 + 5120000);         // 40 KB

    convert_x_kernel  <<<NPTS * CFEAT / 256, 256, 0, stream>>>(x, xb);
    convert_mem_kernel<<<MMEM * CFEAT / 256, 256, 0, stream>>>(memx, mbuf);
    mem_norm_kernel   <<<MMEM, 32, 0, stream>>>(memx, mnorm);

    // 32768 points / (8 waves * 2 tiles * 16 points) = 128 blocks
    knn_wmma_kernel<<<NPTS / 256, 256, 0, stream>>>(xb, mbuf, mnorm, memy, out);
}
